// ViCoTransformer2D_33234456936958
// MI455X (gfx1250) — compile-verified, matrix operand order checked by reading, not run on hardware
//
#include <hip/hip_runtime.h>
#include <math.h>

// ---------------------------------------------------------------------------
// ViCo Transformer2D forward for gfx1250 (MI455X). All GEMMs via
// v_wmma_f32_16x16x32_bf16 (wave32, 16x16 tiles, f32 accumulate).
//
// Every GEMM is full-tile by construction (operands zero-padded where the
// natural shapes are ragged: enc 308->320 rows, dh 40->64, keys 77->80/96),
// so fragment loads are unconditional global_load_b128 + v_cvt_pk_bf16_f32
// with zero per-element bounds logic. Garbage reads past dh=40 on the A side
// are annihilated by zero-padding on the B side.
// ---------------------------------------------------------------------------

typedef __attribute__((ext_vector_type(16))) __bf16 bf16x16;
typedef __attribute__((ext_vector_type(8)))  float  f32x8;

__device__ __forceinline__ f32x8 wmma_bf16(bf16x16 a, bf16x16 b, f32x8 c) {
  return __builtin_amdgcn_wmma_f32_16x16x32_bf16(false, a, false, b,
                                                 (short)0, c, false, false);
}

// ---- A-layout fragment (16x32 tile of row-major source, lane=row) --------
// Lane half 0: K {0..7, 16..23}; half 1: K {8..15, 24..31} (ISA 7.12.2).
// Unconditional: caller guarantees memory validity (16B-aligned, ld%4==0).
__device__ __forceinline__ bf16x16 load_fragA_vec(const float* p, int ld,
                                                  int row0, int k0) {
  int lane = threadIdx.x & 31;
  int hf   = lane >> 4;
  const float* rp = p + (long)(row0 + (lane & 15)) * ld + k0 + hf * 8;
  float4 x0 = *(const float4*)(rp);
  float4 x1 = *(const float4*)(rp + 4);
  float4 y0 = *(const float4*)(rp + 16);
  float4 y1 = *(const float4*)(rp + 20);
  bf16x16 f;
  f[0]  = (__bf16)x0.x; f[1]  = (__bf16)x0.y; f[2]  = (__bf16)x0.z; f[3]  = (__bf16)x0.w;
  f[4]  = (__bf16)x1.x; f[5]  = (__bf16)x1.y; f[6]  = (__bf16)x1.z; f[7]  = (__bf16)x1.w;
  f[8]  = (__bf16)y0.x; f[9]  = (__bf16)y0.y; f[10] = (__bf16)y0.z; f[11] = (__bf16)y0.w;
  f[12] = (__bf16)y1.x; f[13] = (__bf16)y1.y; f[14] = (__bf16)y1.z; f[15] = (__bf16)y1.w;
  return f;
}

// dh=40 high fragment (K 32..63): only lane-half 0 elements 0..7 are real
// (K=32..39); everything else masked to zero. Loads stay inside valid memory.
__device__ __forceinline__ bf16x16 load_frag40_hi(const float* p, int ld, int row0) {
  int lane = threadIdx.x & 31;
  int hf   = lane >> 4;
  const float* rp = p + (long)(row0 + (lane & 15)) * ld + 32 + hf * 8;
  float4 x0 = *(const float4*)(rp);
  float4 x1 = *(const float4*)(rp + 4);
  bool ok = (hf == 0);
  bf16x16 f;
  f[0] = (__bf16)(ok ? x0.x : 0.0f); f[1] = (__bf16)(ok ? x0.y : 0.0f);
  f[2] = (__bf16)(ok ? x0.z : 0.0f); f[3] = (__bf16)(ok ? x0.w : 0.0f);
  f[4] = (__bf16)(ok ? x1.x : 0.0f); f[5] = (__bf16)(ok ? x1.y : 0.0f);
  f[6] = (__bf16)(ok ? x1.z : 0.0f); f[7] = (__bf16)(ok ? x1.w : 0.0f);
#pragma unroll
  for (int e = 8; e < 16; ++e) f[e] = (__bf16)0.0f;
  return f;
}

// ---- B-layout fragment (32x16 tile of row-major K x N source, lane=col) ---
// Unconditional strided loads.
__device__ __forceinline__ bf16x16 load_fragB_raw(const float* p, int ld,
                                                  int k0, int col0) {
  int lane = threadIdx.x & 31;
  int hf   = lane >> 4;
  const float* q  = p + (long)(k0 + hf * 8) * ld + col0 + (lane & 15);
  const float* q2 = q + (long)16 * ld;
  bf16x16 f;
#pragma unroll
  for (int e = 0; e < 8; ++e) {
    f[e]     = (__bf16)q[(long)e * ld];
    f[e + 8] = (__bf16)q2[(long)e * ld];
  }
  return f;
}

// ---------------------------------------------------------------------------
// Batched full-tile GEMM: C = alpha * A*B (+bias[col]) (+R). Wave per 16x16
// tile. Batch decodes (b,h) with separate strides. M,N % 16 == 0, K % 32 == 0,
// all loads memory-valid; only stores are guarded (col < nStore).
// ---------------------------------------------------------------------------
struct GemmP {
  const float* A; int lda; long aB, aH;
  const float* B; int ldb; long bB, bH; int bT;   // bT: B stored N x K
  const float* bias;
  const float* R; int ldr; long rB, rH;
  float* C; int ldc; long cB, cH;
  int M, N, K, nStore; float alpha; int heads;
};

__global__ void gemm_wmma_kernel(GemmP g) {
  int tilesN = g.N >> 4;
  int tiles  = (g.M >> 4) * tilesN;
  int wave   = blockIdx.x * (blockDim.x >> 5) + (threadIdx.x >> 5);
  if (wave >= tiles) return;
  int tm = wave / tilesN, tn = wave - tm * tilesN;
  int bo = blockIdx.y;
  int bb = bo / g.heads, hh = bo - bb * g.heads;
  const float* A = g.A + (long)bb * g.aB + (long)hh * g.aH;
  const float* B = g.B + (long)bb * g.bB + (long)hh * g.bH;
  float*       C = g.C + (long)bb * g.cB + (long)hh * g.cH;

  f32x8 acc;
#pragma unroll
  for (int i = 0; i < 8; ++i) acc[i] = 0.0f;

  if (g.bT) {
    for (int k0 = 0; k0 < g.K; k0 += 32) {
      bf16x16 a = load_fragA_vec(A, g.lda, tm * 16, k0);
      bf16x16 b = load_fragA_vec(B, g.ldb, tn * 16, k0);
      acc = wmma_bf16(a, b, acc);
    }
  } else {
    for (int k0 = 0; k0 < g.K; k0 += 32) {
      bf16x16 a = load_fragA_vec(A, g.lda, tm * 16, k0);
      bf16x16 b = load_fragB_raw(B, g.ldb, k0, tn * 16);
      acc = wmma_bf16(a, b, acc);
    }
  }

  int lane = threadIdx.x & 31, hf = lane >> 4, nl = lane & 15;
  int col = tn * 16 + nl;
  if (col < g.nStore) {
    float bv = g.bias ? g.bias[col] : 0.0f;
    const float* R = g.R ? (g.R + (long)bb * g.rB + (long)hh * g.rH) : nullptr;
#pragma unroll
    for (int i = 0; i < 8; ++i) {
      int row = tm * 16 + i + hf * 8;
      float v = acc[i] * g.alpha + bv;
      if (R) v += R[(long)row * g.ldr + col];
      C[(long)row * g.ldc + col] = v;
    }
  }
}

// ---------------------------------------------------------------------------
// Fused GEGLU GEMM: gg = (A*W[:, :1280]+b[:1280]) * gelu(A*W[:,1280:]+b[1280:])
// ---------------------------------------------------------------------------
__device__ __forceinline__ float gelu_tanh(float x) {
  float x3 = x * x * x;
  return 0.5f * x * (1.0f + tanhf(0.7978845608028654f * (x + 0.044715f * x3)));
}

__global__ void gemm_geglu_kernel(const float* A, int lda, const float* W, int ldw,
                                  const float* bias, float* C, int ldc,
                                  int M, int K) {
  const int N = 1280;
  int tilesN = N >> 4;
  int tiles  = (M >> 4) * tilesN;
  int wave   = blockIdx.x * (blockDim.x >> 5) + (threadIdx.x >> 5);
  if (wave >= tiles) return;
  int tm = wave / tilesN, tn = wave - tm * tilesN;

  f32x8 aca, acg;
#pragma unroll
  for (int i = 0; i < 8; ++i) { aca[i] = 0.0f; acg[i] = 0.0f; }

  for (int k0 = 0; k0 < K; k0 += 32) {
    bf16x16 a  = load_fragA_vec(A, lda, tm * 16, k0);
    bf16x16 b1 = load_fragB_raw(W, ldw, k0, tn * 16);
    bf16x16 b2 = load_fragB_raw(W, ldw, k0, 1280 + tn * 16);
    aca = wmma_bf16(a, b1, aca);
    acg = wmma_bf16(a, b2, acg);
  }

  int lane = threadIdx.x & 31, hf = lane >> 4, nl = lane & 15;
  int col = tn * 16 + nl;
#pragma unroll
  for (int i = 0; i < 8; ++i) {
    int row = tm * 16 + i + hf * 8;
    float va = aca[i] + bias[col];
    float vg = acg[i] + bias[1280 + col];
    C[(long)row * ldc + col] = va * gelu_tanh(vg);
  }
}

// ---------------------------------------------------------------------------
// Fused flash attention (dh=40), wave per 16-query tile, Skv % 32 == 0.
// Q/K/V addressed as row (b*S + s), column (off + h*40 + d).
// Only Q's high fragment masks the dh padding; K-high / V fragments load
// unconditionally (garbage annihilated by Q zeros or lands in unstored output
// columns 40..47). P staged per-wave through LDS in A-frag layout (same-wave
// DS ops are architecturally in-order; no barrier needed).
// ---------------------------------------------------------------------------
__global__ void attn_flash_kernel(const float* Q, int qld, int qoff,
                                  const float* Kp, int kld, int koff,
                                  const float* Vp, int vld, int voff,
                                  float* O, int oldd, int ooff,
                                  const float* bias, int Sq, int Skv,
                                  int heads, float scale) {
  __shared__ __bf16 plds[8][16 * 32];
  int wslot  = threadIdx.x >> 5;
  int qtiles = Sq >> 4;
  int wid = blockIdx.x * 8 + wslot;
  int g   = wid / qtiles, qt = wid - g * qtiles;
  int b   = g / heads,    h  = g - b * heads;
  int lane = threadIdx.x & 31, hf = lane >> 4, nl = lane & 15;

  const float* Qb = Q  + ((long)b * Sq)  * qld + qoff + h * 40;
  const float* Kb = Kp + ((long)b * Skv) * kld + koff + h * 40;
  const float* Vb = Vp + ((long)b * Skv) * vld + voff + h * 40;

  bf16x16 qa0 = load_fragA_vec(Qb, qld, qt * 16, 0);
  bf16x16 qa1 = load_frag40_hi(Qb, qld, qt * 16);

  float m_i[8], l_i[8];
  f32x8 accO[3];
#pragma unroll
  for (int i = 0; i < 8; ++i) {
    m_i[i] = -INFINITY; l_i[i] = 0.0f;
    accO[0][i] = 0.0f; accO[1][i] = 0.0f; accO[2][i] = 0.0f;
  }

  for (int j0 = 0; j0 < Skv; j0 += 32) {
    bf16x16 k0a = load_fragA_vec(Kb, kld, j0,      0);
    bf16x16 k0b = load_fragA_vec(Kb, kld, j0,      32);  // K 40..63 garbage: OK
    bf16x16 k1a = load_fragA_vec(Kb, kld, j0 + 16, 0);
    bf16x16 k1b = load_fragA_vec(Kb, kld, j0 + 16, 32);
    f32x8 s0, s1;
#pragma unroll
    for (int i = 0; i < 8; ++i) { s0[i] = 0.0f; s1[i] = 0.0f; }
    s0 = wmma_bf16(qa0, k0a, s0); s0 = wmma_bf16(qa1, k0b, s0);
    s1 = wmma_bf16(qa0, k1a, s1); s1 = wmma_bf16(qa1, k1b, s1);

    float bi0 = bias ? bias[(long)b * Skv + j0 + nl]      : 0.0f;
    float bi1 = bias ? bias[(long)b * Skv + j0 + 16 + nl] : 0.0f;

    float mnew[8], rs[8];
#pragma unroll
    for (int i = 0; i < 8; ++i) {
      float v0 = s0[i] * scale + bi0;
      float v1 = s1[i] * scale + bi1;
      s0[i] = v0; s1[i] = v1;
      float v = fmaxf(v0, v1);
      for (int off = 1; off < 16; off <<= 1) v = fmaxf(v, __shfl_xor(v, off, 32));
      mnew[i] = fmaxf(m_i[i], v);
    }
#pragma unroll
    for (int i = 0; i < 8; ++i) {
      float p0 = __expf(s0[i] - mnew[i]);
      float p1 = __expf(s1[i] - mnew[i]);
      s0[i] = p0; s1[i] = p1;
      float v = p0 + p1;
      for (int off = 1; off < 16; off <<= 1) v += __shfl_xor(v, off, 32);
      rs[i] = v;
    }
#pragma unroll
    for (int i = 0; i < 8; ++i) {
      float f = __expf(m_i[i] - mnew[i]);
      l_i[i] = l_i[i] * f + rs[i];
      m_i[i] = mnew[i];
      accO[0][i] *= f; accO[1][i] *= f; accO[2][i] *= f;
    }

    // P tile (C-layout) -> LDS (row-major 16x32 bf16) -> A-layout fragment.
#pragma unroll
    for (int i = 0; i < 8; ++i) {
      int row = i + hf * 8;
      plds[wslot][row * 32 + nl]      = (__bf16)s0[i];
      plds[wslot][row * 32 + 16 + nl] = (__bf16)s1[i];
    }
    bf16x16 pf;
    const __bf16* pr = &plds[wslot][(lane & 15) * 32];
#pragma unroll
    for (int e = 0; e < 8; ++e) {
      pf[e]     = pr[hf * 8 + e];
      pf[e + 8] = pr[16 + hf * 8 + e];
    }
#pragma unroll
    for (int t = 0; t < 3; ++t) {
      bf16x16 vb = load_fragB_raw(Vb, vld, j0, t * 16);  // cols 40..47 garbage: OK
      accO[t] = wmma_bf16(pf, vb, accO[t]);
    }
  }

  float* Ob = O + ((long)b * Sq) * oldd + ooff + h * 40;
#pragma unroll
  for (int t = 0; t < 3; ++t) {
    int col = t * 16 + nl;
    if (col < 40) {
#pragma unroll
      for (int i = 0; i < 8; ++i) {
        int row = qt * 16 + i + hf * 8;
        Ob[(long)row * oldd + col] = accO[t][i] / l_i[i];
      }
    }
  }
}

// ---------------------------------------------------------------------------
// Padding / staging kernels (tiny; enable full-tile GEMMs everywhere)
// ---------------------------------------------------------------------------
// enc (308x768) -> encp (320x768, zero rows 308..319)
__global__ void pad_enc_kernel(const float* enc, float* encp) {
  int idx = blockIdx.x * blockDim.x + threadIdx.x;
  if (idx >= 320 * 768) return;
  int r = idx / 768;
  encp[idx] = (r < 308) ? enc[idx] : 0.0f;
}

// kv k-half -> kkp: per (b,h) 80 keys x 64 dh, zero outside (77,40)
__global__ void stage_k_kernel(const float* kv, float* kkp) {
  int idx = blockIdx.x * blockDim.x + threadIdx.x;
  if (idx >= 32 * 80 * 64) return;
  int g = idx / 5120; int rem = idx - g * 5120;
  int n = rem >> 6, d = rem & 63;
  int b = g >> 3, h = g & 7;
  kkp[idx] = (n < 77 && d < 40) ? kv[((long)(b * 77 + n)) * 640 + h * 40 + d] : 0.0f;
}

// kv v-half -> vvp: per (b,h) 96 keys x 48 dh, zero outside (77,40)
__global__ void stage_v_kernel(const float* kv, float* vvp) {
  int idx = blockIdx.x * blockDim.x + threadIdx.x;
  if (idx >= 32 * 96 * 48) return;
  int g = idx / 4608; int rem = idx - g * 4608;
  int k = rem / 48, d = rem - k * 48;
  int b = g >> 3, h = g & 7;
  vvp[idx] = (k < 77 && d < 40) ? kv[((long)(b * 77 + k)) * 640 + 320 + h * 40 + d] : 0.0f;
}

// ---------------------------------------------------------------------------
// Elementwise / normalization / reduction kernels
// ---------------------------------------------------------------------------
__device__ __forceinline__ float bsum(float v, float* red) {
  int tid = threadIdx.x;
  red[tid] = v; __syncthreads();
  for (int s = 128; s > 0; s >>= 1) { if (tid < s) red[tid] += red[tid + s]; __syncthreads(); }
  float r = red[0]; __syncthreads();
  return r;
}
__device__ __forceinline__ float bmax(float v, float* red) {
  int tid = threadIdx.x;
  red[tid] = v; __syncthreads();
  for (int s = 128; s > 0; s >>= 1) { if (tid < s) red[tid] = fmaxf(red[tid], red[tid + s]); __syncthreads(); }
  float r = red[0]; __syncthreads();
  return r;
}
__device__ __forceinline__ float bmin(float v, float* red) {
  int tid = threadIdx.x;
  red[tid] = v; __syncthreads();
  for (int s = 128; s > 0; s >>= 1) { if (tid < s) red[tid] = fminf(red[tid], red[tid + s]); __syncthreads(); }
  float r = red[0]; __syncthreads();
  return r;
}

// GroupNorm (B,C,H,W), 32 groups of 10 ch, eps 1e-6; writes transposed (b,hw,c)
__global__ void groupnorm_kernel(const float* x, const float* w, const float* b,
                                 float* out) {
  __shared__ float red[256];
  int blk = blockIdx.x, bidx = blk / 32, g = blk - bidx * 32;
  const float* xp = x + ((long)bidx * 320 + g * 10) * 1024;
  float s = 0.0f, s2 = 0.0f;
  for (int i = threadIdx.x; i < 10240; i += 256) { float v = xp[i]; s += v; s2 += v * v; }
  s  = bsum(s, red);
  s2 = bsum(s2, red);
  float mu = s / 10240.0f;
  float rstd = rsqrtf(s2 / 10240.0f - mu * mu + 1e-6f);
  for (int i = threadIdx.x; i < 10240; i += 256) {
    int cl = i >> 10, hw = i & 1023, c = g * 10 + cl;
    out[((long)bidx * 1024 + hw) * 320 + c] = (xp[i] - mu) * rstd * w[c] + b[c];
  }
}

// LayerNorm over C=320, one wave per row, eps 1e-5
__global__ void layernorm_kernel(const float* x, const float* w, const float* b,
                                 float* out, int rows) {
  int wid  = (blockIdx.x * blockDim.x + threadIdx.x) >> 5;
  int lane = threadIdx.x & 31;
  if (wid >= rows) return;
  const float* xp = x + (long)wid * 320;
  float s = 0.0f, s2 = 0.0f;
  for (int i = lane; i < 320; i += 32) { float v = xp[i]; s += v; s2 += v * v; }
  for (int off = 16; off > 0; off >>= 1) { s += __shfl_xor(s, off, 32); s2 += __shfl_xor(s2, off, 32); }
  float mu = s / 320.0f;
  float rstd = rsqrtf(s2 / 320.0f - mu * mu + 1e-5f);
  float* op = out + (long)wid * 320;
  for (int i = lane; i < 320; i += 32) op[i] = (xp[i] - mu) * rstd * w[i] + b[i];
}

// In-place row softmax over L valid entries, row stride "stride"
__global__ void softmax_kernel(float* p, int rows, int L, int stride) {
  int wid  = (blockIdx.x * blockDim.x + threadIdx.x) >> 5;
  int lane = threadIdx.x & 31;
  if (wid >= rows) return;
  float* r = p + (long)wid * stride;
  float mx = -INFINITY;
  for (int i = lane; i < L; i += 32) mx = fmaxf(mx, r[i]);
  for (int off = 16; off > 0; off >>= 1) mx = fmaxf(mx, __shfl_xor(mx, off, 32));
  float s = 0.0f;
  for (int i = lane; i < L; i += 32) { float e = __expf(r[i] - mx); r[i] = e; s += e; }
  for (int off = 16; off > 0; off >>= 1) s += __shfl_xor(s, off, 32);
  float inv = 1.0f / s;
  for (int i = lane; i < L; i += 32) r[i] *= inv;
}

// attn_ph / attn_eot gather: mean over heads, pick token column (stride 80)
__global__ void gather_kernel(const float* probs, const int* pb, const int* pt,
                              const int* eb, const int* et, float* ph, float* eot) {
  int tid = blockIdx.x * blockDim.x + threadIdx.x;
  if (tid >= 2048) return;
  int j = tid >> 10, s = tid & 1023;
  {
    int bs = 2 + pb[j]; int t = pt[j];
    float a = 0.0f;
    for (int h = 0; h < 8; ++h) a += probs[(((long)(bs * 8 + h)) * 1024 + s) * 80 + t];
    ph[tid] = a * 0.125f;
  }
  {
    int bs = 2 + eb[j]; int t = et[j];
    float a = 0.0f;
    for (int h = 0; h < 8; ++h) a += probs[(((long)(bs * 8 + h)) * 1024 + s) * 80 + t];
    eot[tid] = a * 0.125f;
  }
}

__global__ void loss_kernel(const float* ph, const float* eot, float* out) {
  __shared__ float red[256];
  int tid = threadIdx.x;
  float mph[2], meo[2];
  for (int j = 0; j < 2; ++j) {
    float m = -INFINITY;
    for (int i = tid; i < 1024; i += 256) m = fmaxf(m, ph[j * 1024 + i]);
    mph[j] = bmax(m, red);
    m = -INFINITY;
    for (int i = tid; i < 1024; i += 256) m = fmaxf(m, eot[j * 1024 + i]);
    meo[j] = bmax(m, red);
  }
  float s = 0.0f;
  for (int i = tid; i < 2048; i += 256) {
    int j = i >> 10;
    float d = ph[i] / mph[j] - eot[i] / meo[j];
    s += d * d;
  }
  s = bsum(s, red);
  if (tid == 0) out[0] = s * (1.0f / 2048.0f);
}

// Otsu threshold -> mask per row (2 blocks, one per bb)
__global__ void otsu_kernel(const float* src, float* mask) {
  __shared__ float red[256];
  __shared__ float s_bestg, s_thr;
  __shared__ int s_best;
  int tid = threadIdx.x;
  const float* m = src + (long)blockIdx.x * 1024;
  float* out = mask + (long)blockIdx.x * 1024;

  float mn = INFINITY, mx = -INFINITY;
  for (int i = tid; i < 1024; i += 256) { float v = m[i]; mn = fminf(mn, v); mx = fmaxf(mx, v); }
  mn = bmin(mn, red);
  mx = bmax(mx, red);
  float iscl = 1.0f / (mx - mn);

  float ts = 0.0f;
  for (int i = tid; i < 1024; i += 256) ts += (m[i] - mn) * iscl;
  float total = bsum(ts, red);

  if (tid == 0) { s_bestg = -INFINITY; s_best = 0; }
  __syncthreads();

  for (int t = 0; t < 10; ++t) {
    float th = 0.1f * t;
    float cl = 0.0f, sl = 0.0f;
    for (int i = tid; i < 1024; i += 256) {
      float xx = (m[i] - mn) * iscl;
      if (xx < th) { cl += 1.0f; sl += xx; }
    }
    float CL = bsum(cl, red);
    float SL = bsum(sl, red);
    if (tid == 0) {
      float CH = 1024.0f - CL;
      float g;
      if (CL <= 0.0f || CH <= 0.0f) g = -INFINITY;
      else {
        float d = SL / CL - (total - SL) / CH;
        g = (CL / 1024.0f) * (CH / 1024.0f) * d * d;
      }
      if (g > s_bestg) { s_bestg = g; s_best = t; }   // first-max like argmax
    }
    __syncthreads();
  }
  if (tid == 0) s_thr = (s_bestg > 0.0f) ? 0.1f * s_best : 0.0f;
  __syncthreads();
  float thr = s_thr;
  for (int i = tid; i < 1024; i += 256) {
    float xx = (m[i] - mn) * iscl;
    out[i] = (xx < thr) ? 0.0f : ((xx > thr) ? 1.0f : xx);
  }
}

// sample[b,c,h,w] = proj_out[(b,hw),c] + residual[b,c,h,w]
__global__ void final_kernel(const float* pw, const float* hid, float* out) {
  int idx = blockIdx.x * blockDim.x + threadIdx.x;
  if (idx >= 1310720) return;
  int b = idx / (320 * 1024);
  int rem = idx - b * 320 * 1024;
  int c = rem >> 10, hw = rem & 1023;
  out[idx] = pw[(((long)b * 1024) + hw) * 320 + c] + hid[idx];
}

// ---------------------------------------------------------------------------
// Host orchestration
// ---------------------------------------------------------------------------
static void launch_gemm(hipStream_t st,
                        const float* A, int lda, const float* B, int ldb, int bT,
                        const float* bias, const float* R, int ldr,
                        float* C, int ldc, int M, int N, int K, int nStore,
                        float alpha,
                        int batches = 1, int heads = 1,
                        long aB = 0, long aH = 0, long bB = 0, long bH = 0,
                        long cB = 0, long cH = 0, long rB = 0, long rH = 0) {
  GemmP g;
  g.A = A; g.lda = lda; g.aB = aB; g.aH = aH;
  g.B = B; g.ldb = ldb; g.bB = bB; g.bH = bH; g.bT = bT;
  g.bias = bias;
  g.R = R; g.ldr = ldr; g.rB = rB; g.rH = rH;
  g.C = C; g.ldc = ldc; g.cB = cB; g.cH = cH;
  g.M = M; g.N = N; g.K = K; g.nStore = nStore; g.alpha = alpha; g.heads = heads;
  int tiles = (M / 16) * (N / 16);
  dim3 grid((tiles + 7) / 8, batches);
  gemm_wmma_kernel<<<grid, 256, 0, st>>>(g);
}

struct BlkW {
  const float *ln1w, *ln1b, *qkv, *ow, *ob, *ln2w, *ln2b, *q2, *kv2,
              *o2w, *o2b, *ln3w, *ln3b, *ff1w, *ff1b, *ff2w, *ff2b;
};

extern "C" void kernel_launch(void* const* d_in, const int* in_sizes, int n_in,
                              void* d_out, int out_size, void* d_ws, size_t ws_size,
                              hipStream_t stream) {
  (void)in_sizes; (void)n_in; (void)out_size; (void)ws_size;
  const float SCALE = 0.15811388300841897f;   // 1/sqrt(40)

  auto F = [&](int i) { return (const float*)d_in[i]; };
  const float* gn_w  = F(0);
  const float* gn_b  = F(1);
  const float* pin_w = F(2);
  const float* pin_b = F(3);
  auto blk = [&](int base) {
    BlkW w;
    w.ln1w = F(base + 0);  w.ln1b = F(base + 1);  w.qkv  = F(base + 2);
    w.ow   = F(base + 3);  w.ob   = F(base + 4);  w.ln2w = F(base + 5);
    w.ln2b = F(base + 6);  w.q2   = F(base + 7);  w.kv2  = F(base + 8);
    w.o2w  = F(base + 9);  w.o2b  = F(base + 10); w.ln3w = F(base + 11);
    w.ln3b = F(base + 12); w.ff1w = F(base + 13); w.ff1b = F(base + 14);
    w.ff2w = F(base + 15); w.ff2b = F(base + 16);
    return w;
  };
  BlkW bw = blk(4);    // b_ block
  BlkW iw = blk(21);   // i_ block
  const float* pout_w = F(38);
  const float* pout_b = F(39);
  const float* hid    = F(40);   // (4,320,32,32)
  const float* enc    = F(41);   // (4,77,768) = 308 x 768
  const int* ph_b = (const int*)d_in[42];
  const int* ph_t = (const int*)d_in[43];
  const int* eot_b = (const int*)d_in[44];
  const int* eot_t = (const int*)d_in[45];

  float* W = (float*)d_ws;
  float* act   = W + 0;          // 4096x320
  float* x     = W + 1310720;    // 4096x320 residual stream
  float* ln    = W + 2621440;    // 4096x320
  float* qkvb  = W + 3932160;    // 4096x960
  float* tmpC  = W + 7864320;    // 4096x320
  float* kv    = W + 9175040;    // 2048x640 (block1 uses 308/320x640 of it)
  float* gg    = W + 10485760;   // 4096x1280
  float* probs = W + 15728640;   // 32 x 1024 x 80 (stride 80, cols>=77 pad)
  float* encp  = W + 18350080;   // 320x768 zero-padded encoder
  float* kkp   = W + 18595840;   // 32 x 80 x 64 zero-padded K heads
  float* vvp   = W + 18759680;   // 32 x 96 x 48 zero-padded V heads
  float* ph    = W + 18907136;   // 2x1024
  float* eot   = W + 18909184;   // 2x1024
  float* mask  = W + 18911232;   // 2x1024
  float* pw    = W + 18913280;   // 4096x320

  float* loss_out = (float*)d_out + 1310720;

  // GroupNorm + transpose to (b, hw, c)
  groupnorm_kernel<<<128, 256, 0, stream>>>(hid, gn_w, gn_b, act);
  // proj_in
  launch_gemm(stream, act, 320, pin_w, 320, 0, pin_b, nullptr, 0, x, 320,
              4096, 320, 320, 320, 1.0f);

  // ---------------- block 1 (b_), all 4 batches, S=1024 -------------------
  layernorm_kernel<<<512, 256, 0, stream>>>(x, bw.ln1w, bw.ln1b, ln, 4096);
  launch_gemm(stream, ln, 320, bw.qkv, 960, 0, nullptr, nullptr, 0, qkvb, 960,
              4096, 960, 320, 960, 1.0f);
  // fused self-attention: 4 batches * 8 heads * 64 q-tiles = 2048 waves
  attn_flash_kernel<<<256, 256, 0, stream>>>(qkvb, 960, 0, qkvb, 960, 320,
                                             qkvb, 960, 640, tmpC, 320, 0,
                                             nullptr, 1024, 1024, 8, SCALE);
  launch_gemm(stream, tmpC, 320, bw.ow, 320, 0, bw.ob, x, 320, x, 320,
              4096, 320, 320, 320, 1.0f);

  layernorm_kernel<<<512, 256, 0, stream>>>(x, bw.ln2w, bw.ln2b, ln, 4096);
  launch_gemm(stream, ln, 320, bw.q2, 320, 0, nullptr, nullptr, 0, tmpC, 320,
              4096, 320, 320, 320, 1.0f);
  // kv = ctx @ kv2 on zero-padded encoder: 320 x 640
  pad_enc_kernel<<<960, 256, 0, stream>>>(enc, encp);
  launch_gemm(stream, encp, 768, bw.kv2, 640, 0, nullptr, nullptr, 0, kv, 640,
              320, 640, 768, 640, 1.0f);
  // stage zero-padded K/V heads
  stage_k_kernel<<<640, 256, 0, stream>>>(kv, kkp);
  stage_v_kernel<<<576, 256, 0, stream>>>(kv, vvp);
  // scores per (b,h): 1024 x 80, K = 64 (A garbage past dh=40 * kkp zeros)
  launch_gemm(stream, tmpC, 320, kkp, 64, 1, nullptr, nullptr, 0, probs, 80,
              1024, 80, 64, 77, SCALE, 32, 8,
              /*aB*/ 1024L * 320, /*aH*/ 40, /*bB*/ 8L * 80 * 64, /*bH*/ 80L * 64,
              /*cB*/ 8L * 1024 * 80, /*cH*/ 1024L * 80, 0, 0);
  softmax_kernel<<<4096, 256, 0, stream>>>(probs, 32768, 77, 80);
  // O = P @ V per (b,h): 1024 x 48, K = 96 (A garbage cols * vvp zero rows)
  launch_gemm(stream, probs, 80, vvp, 48, 0, nullptr, nullptr, 0, tmpC, 320,
              1024, 48, 96, 40, 1.0f, 32, 8,
              /*aB*/ 8L * 1024 * 80, /*aH*/ 1024L * 80, /*bB*/ 8L * 96 * 48, /*bH*/ 96L * 48,
              /*cB*/ 1024L * 320, /*cH*/ 40, 0, 0);
  launch_gemm(stream, tmpC, 320, bw.o2w, 320, 0, bw.o2b, x, 320, x, 320,
              4096, 320, 320, 320, 1.0f);

  layernorm_kernel<<<512, 256, 0, stream>>>(x, bw.ln3w, bw.ln3b, ln, 4096);
  gemm_geglu_kernel<<<2560, 256, 0, stream>>>(ln, 320, bw.ff1w, 2560, bw.ff1b,
                                              gg, 1280, 4096, 320);
  launch_gemm(stream, gg, 1280, bw.ff2w, 320, 0, bw.ff2b, x, 320, x, 320,
              4096, 320, 1280, 320, 1.0f);

  // ---------------- attention stats: loss + otsu mask ---------------------
  gather_kernel<<<8, 256, 0, stream>>>(probs, ph_b, ph_t, eot_b, eot_t, ph, eot);
  loss_kernel<<<1, 256, 0, stream>>>(ph, eot, loss_out);
  otsu_kernel<<<2, 256, 0, stream>>>(ph, mask);

  // ---------------- block 2 (i_), first 2 batches; ctx = img half ---------
  layernorm_kernel<<<256, 256, 0, stream>>>(x, iw.ln1w, iw.ln1b, ln, 2048);
  launch_gemm(stream, ln, 320, iw.qkv, 960, 0, nullptr, nullptr, 0, qkvb, 960,
              2048, 960, 320, 960, 1.0f);
  attn_flash_kernel<<<128, 256, 0, stream>>>(qkvb, 960, 0, qkvb, 960, 320,
                                             qkvb, 960, 640, tmpC, 320, 0,
                                             nullptr, 1024, 1024, 8, SCALE);
  launch_gemm(stream, tmpC, 320, iw.ow, 320, 0, iw.ob, x, 320, x, 320,
              2048, 320, 320, 320, 1.0f);

  layernorm_kernel<<<256, 256, 0, stream>>>(x, iw.ln2w, iw.ln2b, ln, 2048);
  launch_gemm(stream, ln, 320, iw.q2, 320, 0, nullptr, nullptr, 0, tmpC, 320,
              2048, 320, 320, 320, 1.0f);
  // kv = img @ i_kv2, img = x rows 2048..4095 (untouched by block 2)
  launch_gemm(stream, x + 2048L * 320, 320, iw.kv2, 640, 0, nullptr, nullptr, 0,
              kv, 640, 2048, 640, 320, 640, 1.0f);
  // fused cross-attention with otsu-mask bias over 1024 keys
  attn_flash_kernel<<<128, 256, 0, stream>>>(tmpC, 320, 0, kv, 640, 0,
                                             kv, 640, 320, ln, 320, 0,
                                             mask, 1024, 1024, 8, SCALE);
  launch_gemm(stream, ln, 320, iw.o2w, 320, 0, iw.o2b, x, 320, x, 320,
              2048, 320, 320, 320, 1.0f);

  layernorm_kernel<<<256, 256, 0, stream>>>(x, iw.ln3w, iw.ln3b, ln, 2048);
  gemm_geglu_kernel<<<1280, 256, 0, stream>>>(ln, 320, iw.ff1w, 2560, iw.ff1b,
                                              gg, 1280, 2048, 320);
  launch_gemm(stream, gg, 1280, iw.ff2w, 320, 0, iw.ff2b, x, 320, x, 320,
              2048, 320, 1280, 320, 1.0f);

  // ---------------- proj_out + residual + layout --------------------------
  launch_gemm(stream, x, 320, pout_w, 320, 0, pout_b, nullptr, 0, pw, 320,
              4096, 320, 320, 320, 1.0f);
  final_kernel<<<5120, 256, 0, stream>>>(pw, hid, (float*)d_out);
}
